// VectorQuantizer_62792421867639
// MI455X (gfx1250) — matrix operand order repaired and checked by reading, not compiled, not run on hardware
//
#include <hip/hip_runtime.h>

typedef __attribute__((ext_vector_type(16))) __bf16 v16bf;
typedef __attribute__((ext_vector_type(8)))  float  v8f;

#define B_ROWS  32768
#define K_CODES 8192
#define D_DIM   256
#define EPAD    264   // 256 + 8 bf16 pad -> 132 dwords/row -> conflict-free b128 LDS reads
#define NSTAGES (K_CODES / 32)

__device__ __forceinline__ unsigned short f32_to_bf16(float f) {
  unsigned u = __float_as_uint(f);
  unsigned r = u + 0x7FFFu + ((u >> 16) & 1u);   // round-to-nearest-even
  return (unsigned short)(r >> 16);
}
__device__ __forceinline__ float bf16_to_f32(unsigned short h) {
  return __uint_as_float(((unsigned)h) << 16);
}

// ---------------------------------------------------------------------------
// Kernel 1: split codebook E (K x D, f32) into bf16 hi/lo planes + row norms.
// ---------------------------------------------------------------------------
__global__ __launch_bounds__(256) void vq_prep(const float* __restrict__ E,
                                               unsigned short* __restrict__ Ehi,
                                               unsigned short* __restrict__ Elo,
                                               float* __restrict__ enorm) {
  const int row = blockIdx.x;
  const int t   = threadIdx.x;
  const size_t g = (size_t)row * D_DIM + t;
  const float x = E[g];
  const unsigned short h = f32_to_bf16(x);
  Ehi[g] = h;
  Elo[g] = f32_to_bf16(x - bf16_to_f32(h));

  float s = x * x;
  #pragma unroll
  for (int o = 16; o > 0; o >>= 1) s += __shfl_down(s, o, 32);
  __shared__ float sm[8];
  const int lane = t & 31, w = t >> 5;
  if (lane == 0) sm[w] = s;
  __syncthreads();
  if (w == 0) {
    float v = (lane < 8) ? sm[lane] : 0.f;
    #pragma unroll
    for (int o = 4; o > 0; o >>= 1) v += __shfl_down(v, o, 32);
    if (lane == 0) enorm[row] = v;
  }
}

// ---------------------------------------------------------------------------
// Kernel 2: argmin_k ( ||e_k||^2 - 2 * z . e_k )  via bf16x2-split WMMA.
// 8 waves/WG, each wave owns 16 z rows (A frags resident in VGPRs).
// E is double-buffered in LDS and streamed with GLOBAL_LOAD_ASYNC_TO_LDS_B128,
// overlapping the L2 stream (ASYNCcnt) with the WMMA stream.
// ---------------------------------------------------------------------------
__global__ __launch_bounds__(256) void vq_argmin(const float* __restrict__ z,
                                                 const unsigned short* __restrict__ Ehi,
                                                 const unsigned short* __restrict__ Elo,
                                                 const float* __restrict__ enorm,
                                                 unsigned int* __restrict__ idx_out) {
  __shared__ __attribute__((aligned(16))) unsigned short eh[2][32][EPAD];
  __shared__ __attribute__((aligned(16))) unsigned short el[2][32][EPAD];

  const int tid   = threadIdx.x;
  const int lane  = tid & 31;
  const int wave  = tid >> 5;
  const int lhalf = lane >> 4;      // 0: lanes 0-15, 1: lanes 16-31
  const int lm    = lane & 15;

  // Per-thread slice of one 32-codeword stage: 32 bf16 per plane (4 x b128).
  const int rl = tid >> 3;          // codeword row within stage
  const int c0 = (tid & 7) * 32;    // column start
  const unsigned long long gbh = (unsigned long long)(uintptr_t)Ehi;
  const unsigned long long gbl = (unsigned long long)(uintptr_t)Elo;

  auto issue_stage = [&](int stage, int buf) {
    const unsigned goff = (unsigned)(((unsigned)(stage * 32 + rl) * D_DIM + c0) * 2u);
    #pragma unroll
    for (int q = 0; q < 4; ++q) {
      const unsigned dh = (unsigned)(uintptr_t)&eh[buf][rl][c0 + q * 8];
      const unsigned dl = (unsigned)(uintptr_t)&el[buf][rl][c0 + q * 8];
      asm volatile("global_load_async_to_lds_b128 %0, %1, %2"
                   :: "v"(dh), "v"(goff + q * 16u), "s"(gbh) : "memory");
      asm volatile("global_load_async_to_lds_b128 %0, %1, %2"
                   :: "v"(dl), "v"(goff + q * 16u), "s"(gbl) : "memory");
    }
  };

  // ---- Preload this wave's 16 z rows as bf16 hi/lo A-fragments (ISA 7.12.2
  // 16-bit A 16x32: lanes 0-15 row=lm hold K {0-7,16-23}; lanes 16-31 {8-15,24-31}).
  const int rowg = blockIdx.x * 128 + wave * 16 + lm;
  const float* zrow = z + (size_t)rowg * D_DIM;

  // Kick off stage 0 while we convert A fragments.
  issue_stage(0, 0);

  v16bf ahi[8], alo[8];
  #pragma unroll
  for (int c = 0; c < 8; ++c) {
    const int k0 = 32 * c + 8 * lhalf;
    float f[16];
    *(float4*)&f[0]  = *(const float4*)(zrow + k0);
    *(float4*)&f[4]  = *(const float4*)(zrow + k0 + 4);
    *(float4*)&f[8]  = *(const float4*)(zrow + k0 + 16);
    *(float4*)&f[12] = *(const float4*)(zrow + k0 + 20);
    union { v16bf v; unsigned short s[16]; } H, L;
    #pragma unroll
    for (int i = 0; i < 16; ++i) {
      const unsigned short h = f32_to_bf16(f[i]);
      H.s[i] = h;
      L.s[i] = f32_to_bf16(f[i] - bf16_to_f32(h));
    }
    ahi[c] = H.v;
    alo[c] = L.v;
  }

  float mv[8];
  int   mi[8];
  #pragma unroll
  for (int i = 0; i < 8; ++i) { mv[i] = 3.4e38f; mi[i] = 0; }

  asm volatile("s_wait_asynccnt 0x0" ::: "memory");
  __syncthreads();

  for (int nb = 0; nb < NSTAGES; ++nb) {
    const int cur = nb & 1;
    // Stream the next stage into the other buffer while we compute on this one.
    if (nb + 1 < NSTAGES) issue_stage(nb + 1, cur ^ 1);

    #pragma unroll
    for (int s = 0; s < 2; ++s) {
      const int   nl    = s * 16 + lm;        // codeword within stage (B column)
      const int   nglob = nb * 32 + nl;
      const float en    = enorm[nglob];

      v8f hh = {0,0,0,0,0,0,0,0};
      v8f hl = {0,0,0,0,0,0,0,0};
      v8f lh = {0,0,0,0,0,0,0,0};
      #pragma unroll
      for (int c = 0; c < 8; ++c) {
        // B 32x16 bf16: lanes 0-15 hold K {0-7,16-23}, lanes 16-31 hold K {8-15,24-31}.
        const int kk = 32 * c + 8 * lhalf;
        union { v16bf v; uint4 q[2]; } Bh, Bl;
        Bh.q[0] = *(const uint4*)&eh[cur][nl][kk];
        Bh.q[1] = *(const uint4*)&eh[cur][nl][kk + 16];
        Bl.q[0] = *(const uint4*)&el[cur][nl][kk];
        Bl.q[1] = *(const uint4*)&el[cur][nl][kk + 16];
        hh = __builtin_amdgcn_wmma_f32_16x16x32_bf16(false, ahi[c], false, Bh.v, (short)0, hh, false, false);
        hl = __builtin_amdgcn_wmma_f32_16x16x32_bf16(false, ahi[c], false, Bl.v, (short)0, hl, false, false);
        lh = __builtin_amdgcn_wmma_f32_16x16x32_bf16(false, alo[c], false, Bh.v, (short)0, lh, false, false);
      }
      // C layout: vgpr i, this lane -> row (i + 8*lhalf), col nglob.
      #pragma unroll
      for (int i = 0; i < 8; ++i) {
        const float dist = en - 2.0f * (hh[i] + hl[i] + lh[i]);
        if (dist < mv[i]) { mv[i] = dist; mi[i] = nglob; }  // ascending n => first-min kept
      }
    }

    // All async writes into buf cur^1 done (per-wave), then WG-wide hand-off.
    asm volatile("s_wait_asynccnt 0x0" ::: "memory");
    __syncthreads();
  }

  // ---- Cross-lane argmin within each 16-lane half (same 8 rows per half).
  #pragma unroll
  for (int i = 0; i < 8; ++i) {
    float v  = mv[i];
    int   ix = mi[i];
    #pragma unroll
    for (int o = 1; o < 16; o <<= 1) {
      const float ov = __shfl_xor(v, o, 32);
      const int   oi = __shfl_xor(ix, o, 32);
      if (ov < v || (ov == v && oi < ix)) { v = ov; ix = oi; }
    }
    if (lm == 0)
      idx_out[blockIdx.x * 128 + wave * 16 + lhalf * 8 + i] = (unsigned)ix;
  }
}

// ---------------------------------------------------------------------------
// Kernel 3: z_q gather (== forward value of the STE output) + per-row partial
// sums of (z_q - z)^2 (deterministic fixed-order reduction, no atomics).
// ---------------------------------------------------------------------------
__global__ __launch_bounds__(256) void vq_gather(const float* __restrict__ z,
                                                 const float* __restrict__ E,
                                                 const unsigned int* __restrict__ idx,
                                                 float* __restrict__ out,
                                                 float* __restrict__ partial) {
  const int row = blockIdx.x;
  const int t   = threadIdx.x;
  const unsigned k = idx[row];
  const float q = E[(size_t)k * D_DIM + t];
  const size_t o = (size_t)row * D_DIM + t;
  out[o] = q;
  const float d = q - z[o];
  float s = d * d;
  #pragma unroll
  for (int off = 16; off > 0; off >>= 1) s += __shfl_down(s, off, 32);
  __shared__ float sm[8];
  const int lane = t & 31, w = t >> 5;
  if (lane == 0) sm[w] = s;
  __syncthreads();
  if (w == 0) {
    float v = (lane < 8) ? sm[lane] : 0.f;
    #pragma unroll
    for (int off = 4; off > 0; off >>= 1) v += __shfl_down(v, off, 32);
    if (lane == 0) partial[row] = v;
  }
}

// Kernel 4: single block folds the 32768 row partials -> vq_loss scalar.
__global__ __launch_bounds__(256) void vq_final(const float* __restrict__ partial,
                                                float* __restrict__ out) {
  float s = 0.f;
  for (int i = threadIdx.x; i < B_ROWS; i += 256) s += partial[i];
  #pragma unroll
  for (int off = 16; off > 0; off >>= 1) s += __shfl_down(s, off, 32);
  __shared__ float sm[8];
  const int lane = threadIdx.x & 31, w = threadIdx.x >> 5;
  if (lane == 0) sm[w] = s;
  __syncthreads();
  if (w == 0) {
    float v = (lane < 8) ? sm[lane] : 0.f;
    #pragma unroll
    for (int off = 4; off > 0; off >>= 1) v += __shfl_down(v, off, 32);
    if (lane == 0)
      out[(size_t)B_ROWS * D_DIM] = v * (1.25f / (float)((size_t)B_ROWS * D_DIM));
  }
}

// ---------------------------------------------------------------------------
extern "C" void kernel_launch(void* const* d_in, const int* in_sizes, int n_in,
                              void* d_out, int out_size, void* d_ws, size_t ws_size,
                              hipStream_t stream) {
  const float* z = (const float*)d_in[0];          // (B, D) f32
  const float* E = (const float*)d_in[1];          // (K, D) f32
  float* out = (float*)d_out;                      // B*D z_q_ste + 1 loss

  // Workspace layout (~8.7 MB total)
  unsigned short* Ehi   = (unsigned short*)d_ws;                    // K*D bf16
  unsigned short* Elo   = Ehi + (size_t)K_CODES * D_DIM;            // K*D bf16
  float*          enorm = (float*)(Elo + (size_t)K_CODES * D_DIM);  // K f32
  unsigned int*   idx   = (unsigned int*)(enorm + K_CODES);         // B u32
  float*          part  = (float*)(idx + B_ROWS);                   // B f32

  vq_prep  <<<K_CODES,       256, 0, stream>>>(E, Ehi, Elo, enorm);
  vq_argmin<<<B_ROWS / 128,  256, 0, stream>>>(z, Ehi, Elo, enorm, idx);
  vq_gather<<<B_ROWS,        256, 0, stream>>>(z, E, idx, out, part);
  vq_final <<<1,             256, 0, stream>>>(part, out);
}
